// MHA_15229954031990
// MI455X (gfx1250) — matrix operand order repaired and checked by reading, not compiled, not run on hardware
//
#include <hip/hip_runtime.h>

// ---------------------------------------------------------------------------
// Causal MHA block for MI455X (gfx1250, wave32, WMMA bf16 16x16x32, f32 acc)
// ---------------------------------------------------------------------------
#define Bc 4
#define Tc 2048
#define Cc 1024
#define Hc 16
#define Dc 64
#define C3 (3 * Cc)
#define BT (Bc * Tc)         // 8192
#define BH (Bc * Hc)         // 64
#define SCALE 0.125f         // 64^-0.5

typedef __attribute__((ext_vector_type(16))) __bf16 v16bf;
typedef __attribute__((ext_vector_type(8)))  __bf16 v8bf;
typedef __attribute__((ext_vector_type(8)))  float  v8f;

union V16U { v16bf v; v8bf h[2]; };

// A-fragment (16x32 bf16, MxK): lane = row M (mod 16); two 16B segments:
//   e=0..7  -> K = kbase + (lane>=16?16:0) + e
//   e=8..15 -> K = kbase + (lane>=16?16:0) + e
__device__ inline v16bf load_afrag(const __bf16* __restrict__ row, int kbase, int lane) {
  const int s = (lane >= 16) ? 16 : 0;
  V16U u;
  u.h[0] = *(const v8bf*)(row + kbase + s);
  u.h[1] = *(const v8bf*)(row + kbase + s + 8);
  return u.v;
}

__device__ inline v8f wmma_bf16(v16bf a, v16bf b, v8f c) {
  return __builtin_amdgcn_wmma_f32_16x16x32_bf16(false, a, false, b, (short)0, c, false, false);
}

// ---------------------------------------------------------------------------
// Prologue casts
// ---------------------------------------------------------------------------
__global__ void __launch_bounds__(256) k_cast_bf16(const float* __restrict__ src,
                                                   __bf16* __restrict__ dst, int n) {
  int i = blockIdx.x * 256 + threadIdx.x;
  if (i < n) dst[i] = (__bf16)src[i];
}

// dst[n*K + k] = src[k*N + n]   (src: [K][N] f32, dst: [N][K] bf16)
__global__ void __launch_bounds__(256) k_transpose_bf16(const float* __restrict__ src,
                                                        __bf16* __restrict__ dst,
                                                        int N, int K) {
  int i = blockIdx.x * 256 + threadIdx.x;
  if (i >= N * K) return;
  int n = i / K, k = i - n * K;
  dst[i] = (__bf16)src[(size_t)k * N + n];
}

// ---------------------------------------------------------------------------
// QKV GEMM: qkv[m, n] = sum_k x[m,k] * w_qkv[k,n];   M=8192, K=1024, N=3072.
// Wave computes a 32x64 tile (2 A-frags x 4 B-frags = 8 WMMA / K-step; every
// B-frag reused twice in registers). Each 64-col group == one head's full D.
// Scatter: Q,K -> [BH, T, 64] bf16; V -> [BH, 64, T] bf16 (transposed).
// ---------------------------------------------------------------------------
__global__ void __launch_bounds__(256) k_qkv_gemm(const __bf16* __restrict__ xb,
                                                  const __bf16* __restrict__ wT,
                                                  __bf16* __restrict__ qb,
                                                  __bf16* __restrict__ kb,
                                                  __bf16* __restrict__ vT) {
  const int lane = threadIdx.x & 31;
  const int w    = threadIdx.x >> 5;
  const int m0   = (blockIdx.x * 8 + w) * 32;
  const int n0   = blockIdx.y * 64;
  const int rsel = lane & 15;
  const int bsh  = (lane >= 16) ? 16 : 0;

  const __bf16* arow0 = xb + (size_t)(m0 + rsel) * Cc;
  const __bf16* arow1 = arow0 + (size_t)16 * Cc;
  v8f acc[2][4] = {};
  for (int k = 0; k < Cc; k += 32) {
    v16bf a0 = load_afrag(arow0, k, lane);
    v16bf a1 = load_afrag(arow1, k, lane);
#pragma unroll
    for (int nt = 0; nt < 4; ++nt) {
      const __bf16* bp = wT + (size_t)(n0 + nt * 16 + rsel) * Cc + k + bsh;
      v16bf b = *(const v16bf*)bp;
      acc[0][nt] = wmma_bf16(a0, b, acc[0][nt]);
      acc[1][nt] = wmma_bf16(a1, b, acc[1][nt]);
    }
  }

  const int seg = n0 >> 10;            // 0=q, 1=k, 2=v  (uniform per block)
  const int hh  = (n0 & 1023) >> 6;    // head index
  const int Mb  = (lane >= 16) ? 8 : 0;
#pragma unroll
  for (int mt = 0; mt < 2; ++mt) {
#pragma unroll
    for (int nt = 0; nt < 4; ++nt) {
#pragma unroll
      for (int r = 0; r < 8; ++r) {
        int m = m0 + mt * 16 + Mb + r;
        int d = nt * 16 + rsel;
        int bi = m >> 11, t = m & (Tc - 1);
        size_t bh = (size_t)bi * Hc + hh;
        __bf16 val = (__bf16)acc[mt][nt][r];
        if (seg == 0)      qb[(bh * Tc + t) * Dc + d] = val;
        else if (seg == 1) kb[(bh * Tc + t) * Dc + d] = val;
        else               vT[(bh * Dc + d) * Tc + t] = val;
      }
    }
  }
}

// ---------------------------------------------------------------------------
// Flash attention: one wave per (bh, 16-query tile); stream 32 keys per chunk.
// ---------------------------------------------------------------------------
__global__ void __launch_bounds__(128) k_attn(const __bf16* __restrict__ qb,
                                              const __bf16* __restrict__ kb,
                                              const __bf16* __restrict__ vT,
                                              __bf16* __restrict__ yb) {
  __shared__ __align__(32) __bf16 lds_p[4][16 * 32];
  const int lane = threadIdx.x & 31;
  const int w    = threadIdx.x >> 5;
  const int tile = blockIdx.x * 4 + w;
  const int bh   = tile >> 7;            // / 128 q-tiles
  const int qt   = tile & 127;
  const int q0   = qt * 16;
  const int row  = lane & 15;            // N (C-layout) / M (A-layout)
  const int Mb   = (lane >= 16) ? 8 : 0;
  const int s8   = Mb;                   // unused; A-frag K sub-offset is bsh
  const int bsh  = (lane >= 16) ? 16 : 0;

  const __bf16* qrow = qb + ((size_t)bh * Tc + q0 + row) * Dc;
  v16bf qf0 = load_afrag(qrow, 0, lane);
  v16bf qf1 = load_afrag(qrow, 32, lane);

  v8f o[4] = {};
  float mrow[8], lrow[8];
#pragma unroll
  for (int r = 0; r < 8; ++r) { mrow[r] = -1e30f; lrow[r] = 0.0f; }

  __bf16* lp = lds_p[w];
  const int nchunk = (q0 + 15) / 32 + 1;

  for (int j = 0; j < nchunk; ++j) {
    const int cb = j * 32;
    // ---- S = Q @ K^T for keys [cb, cb+32) ----
    const __bf16* kr0 = kb + ((size_t)bh * Tc + cb + row) * Dc + bsh;
    const __bf16* kr1 = kr0 + 16 * Dc;
    v8f s0 = {}, s1 = {};
    s0 = wmma_bf16(qf0, *(const v16bf*)(kr0),      s0);
    s0 = wmma_bf16(qf1, *(const v16bf*)(kr0 + 32), s0);
    s1 = wmma_bf16(qf0, *(const v16bf*)(kr1),      s1);
    s1 = wmma_bf16(qf1, *(const v16bf*)(kr1 + 32), s1);

    // ---- scale + causal mask (only the boundary chunk can mask) ----
#pragma unroll
    for (int r = 0; r < 8; ++r) { s0[r] *= SCALE; s1[r] *= SCALE; }
    if (cb + 32 > q0) {
#pragma unroll
      for (int r = 0; r < 8; ++r) {
        int qg = q0 + Mb + r;
        if (cb + row      > qg) s0[r] = -1e30f;
        if (cb + 16 + row > qg) s1[r] = -1e30f;
      }
    }

    // ---- online softmax (row reductions across 16-lane halves) ----
#pragma unroll
    for (int r = 0; r < 8; ++r) {
      float mx = fmaxf(s0[r], s1[r]);
#pragma unroll
      for (int sh = 8; sh >= 1; sh >>= 1) mx = fmaxf(mx, __shfl_xor(mx, sh));
      float mnew  = fmaxf(mrow[r], mx);
      float alpha = __expf(mrow[r] - mnew);
      mrow[r] = mnew;
      float p0 = __expf(s0[r] - mnew);
      float p1 = __expf(s1[r] - mnew);
      s0[r] = p0; s1[r] = p1;
      float ps = p0 + p1;
#pragma unroll
      for (int sh = 8; sh >= 1; sh >>= 1) ps += __shfl_xor(ps, sh);
      lrow[r] = lrow[r] * alpha + ps;
      o[0][r] *= alpha; o[1][r] *= alpha; o[2][r] *= alpha; o[3][r] *= alpha;
    }

    // ---- C-frag -> A-frag relayout of P through per-wave LDS ----
#pragma unroll
    for (int r = 0; r < 8; ++r) {
      int M = Mb + r;
      lp[M * 32 + row]      = (__bf16)s0[r];
      lp[M * 32 + 16 + row] = (__bf16)s1[r];
    }
    V16U pu;
    pu.h[0] = *(const v8bf*)(lp + row * 32 + bsh);
    pu.h[1] = *(const v8bf*)(lp + row * 32 + bsh + 8);

    // ---- O += P @ V ----
#pragma unroll
    for (int nt = 0; nt < 4; ++nt) {
      const __bf16* vp = vT + ((size_t)bh * Dc + nt * 16 + row) * Tc + cb + bsh;
      o[nt] = wmma_bf16(pu.v, *(const v16bf*)vp, o[nt]);
    }
  }

  // ---- normalize and write y (bf16, [B,T,C] row-major) ----
  const int bi = bh >> 4, hh = bh & 15;
#pragma unroll
  for (int r = 0; r < 8; ++r) {
    float inv = 1.0f / lrow[r];
    int t = q0 + Mb + r;
    size_t base = ((size_t)bi * Tc + t) * Cc + hh * Dc;
#pragma unroll
    for (int nt = 0; nt < 4; ++nt)
      yb[base + nt * 16 + row] = (__bf16)(o[nt][r] * inv);
  }
}

// ---------------------------------------------------------------------------
// Output projection: out = y @ w_proj; M=8192, K=1024, N=1024, f32 out.
// Wave computes a 32x64 tile (same register-reuse scheme as QKV GEMM).
// ---------------------------------------------------------------------------
__global__ void __launch_bounds__(256) k_proj_gemm(const __bf16* __restrict__ yb,
                                                   const __bf16* __restrict__ wT,
                                                   float* __restrict__ out) {
  const int lane = threadIdx.x & 31;
  const int w    = threadIdx.x >> 5;
  const int m0   = (blockIdx.x * 8 + w) * 32;
  const int n0   = blockIdx.y * 64;
  const int rsel = lane & 15;
  const int bsh  = (lane >= 16) ? 16 : 0;

  const __bf16* arow0 = yb + (size_t)(m0 + rsel) * Cc;
  const __bf16* arow1 = arow0 + (size_t)16 * Cc;
  v8f acc[2][4] = {};
  for (int k = 0; k < Cc; k += 32) {
    v16bf a0 = load_afrag(arow0, k, lane);
    v16bf a1 = load_afrag(arow1, k, lane);
#pragma unroll
    for (int nt = 0; nt < 4; ++nt) {
      const __bf16* bp = wT + (size_t)(n0 + nt * 16 + rsel) * Cc + k + bsh;
      v16bf b = *(const v16bf*)bp;
      acc[0][nt] = wmma_bf16(a0, b, acc[0][nt]);
      acc[1][nt] = wmma_bf16(a1, b, acc[1][nt]);
    }
  }
  const int Mb = (lane >= 16) ? 8 : 0;
#pragma unroll
  for (int mt = 0; mt < 2; ++mt)
#pragma unroll
    for (int nt = 0; nt < 4; ++nt)
#pragma unroll
      for (int r = 0; r < 8; ++r)
        out[(size_t)(m0 + mt * 16 + Mb + r) * Cc + n0 + nt * 16 + rsel] = acc[mt][nt][r];
}

// ---------------------------------------------------------------------------
extern "C" void kernel_launch(void* const* d_in, const int* in_sizes, int n_in,
                              void* d_out, int out_size, void* d_ws, size_t ws_size,
                              hipStream_t stream) {
  const float* x      = (const float*)d_in[0];   // [B,T,C]
  const float* w_qkv  = (const float*)d_in[1];   // [C,3C]
  const float* w_proj = (const float*)d_in[2];   // [C,C]
  float* out          = (float*)d_out;           // [B,T,C]

  char* ws = (char*)d_ws;
  size_t o_xb    = 0;
  size_t o_wqkvT = o_xb    + (size_t)BT * Cc * 2;   // 16 MB
  size_t o_wprjT = o_wqkvT + (size_t)C3 * Cc * 2;   //  6 MB
  size_t o_q     = o_wprjT + (size_t)Cc * Cc * 2;   //  2 MB
  size_t o_k     = o_q     + (size_t)BT * Cc * 2;   // 16 MB
  size_t o_v     = o_k     + (size_t)BT * Cc * 2;   // 16 MB
  size_t o_y     = o_v     + (size_t)BT * Cc * 2;   // 16 MB
  __bf16* xb    = (__bf16*)(ws + o_xb);
  __bf16* wqkvT = (__bf16*)(ws + o_wqkvT);
  __bf16* wprjT = (__bf16*)(ws + o_wprjT);
  __bf16* qb    = (__bf16*)(ws + o_q);
  __bf16* kb    = (__bf16*)(ws + o_k);
  __bf16* vT    = (__bf16*)(ws + o_v);
  __bf16* yb    = (__bf16*)(ws + o_y);

  // 1) casts / weight transposes
  k_cast_bf16<<<(BT * Cc) / 256, 256, 0, stream>>>(x, xb, BT * Cc);
  k_transpose_bf16<<<(C3 * Cc) / 256, 256, 0, stream>>>(w_qkv, wqkvT, C3, Cc);
  k_transpose_bf16<<<(Cc * Cc) / 256, 256, 0, stream>>>(w_proj, wprjT, Cc, Cc);

  // 2) QKV projection (WMMA), scattering into attention layouts
  k_qkv_gemm<<<dim3(BT / 256, C3 / 64), 256, 0, stream>>>(xb, wqkvT, qb, kb, vT);

  // 3) flash attention (WMMA + online softmax)
  k_attn<<<(BH * 128) / 4, 128, 0, stream>>>(qb, kb, vT, yb);

  // 4) output projection (WMMA) -> f32 out
  k_proj_gemm<<<dim3(BT / 256, Cc / 64), 256, 0, stream>>>(yb, wprjT, out);
}
